// Seq2seqAttn_86414741996274
// MI455X (gfx1250) — compile-verified, matrix operand order checked by reading, not run on hardware
//
#include <hip/hip_runtime.h>

// ---------------------------------------------------------------------------
// Types for CDNA5 WMMA
// ---------------------------------------------------------------------------
typedef _Float16 h16;
typedef __attribute__((ext_vector_type(16))) _Float16 v16h;
typedef __attribute__((ext_vector_type(8)))  float    v8f;
typedef unsigned int v4u __attribute__((ext_vector_type(4)));

union FragU { v16h v; v4u q[2]; };

// Model dims
#define BZ   64
#define LS   512
#define LL   128
#define TT   150
#define HH   150
#define G3   450      // 3*H
#define EM   200
#define EP   224      // E padded to mult of 32
#define HP   160      // H padded to mult of 32
#define VT   10000
#define NC   62
#define SOS_TOK 2
#define EOS_TOK 1

// ---------------------------------------------------------------------------
// Generic batched WMMA GEMM:  C[b] = A[b] (MxK,f16,row) * B[b]^T (NxK,f16,row)
//                                    (+bias[n]) (+residual) (tanh?) (row remap?)
// blockDim = (32,4): 4 waves per block, each wave owns one 16x16 C tile
// (same M-tile, 4 consecutive N-tiles).  K must be a multiple of 32.
// A/B fragment loads are unconditional b128 loads from row-clamped addresses:
// out-of-range rows only feed never-stored outputs, so no predication needed.
// ---------------------------------------------------------------------------
__global__ void gemm_wmma_kernel(
    const h16* __restrict__ A, const h16* __restrict__ Bm,
    const float* __restrict__ bias, const float* __restrict__ residual,
    float* __restrict__ C,
    int M, int N, int K, int lda, int ldb, int ldc,
    long aStride, long bStride, long cStride, long rStride,
    int doTanh, int remapT, int remapB)
{
  int lane = threadIdx.x;                       // 0..31
  int tn = blockIdx.x * 4 + threadIdx.y;        // N tile
  int tm = blockIdx.y;                          // M tile
  int bz = blockIdx.z;
  const h16* Ab = A + (long)bz * aStride;
  const h16* Bb = Bm + (long)bz * bStride;
  float*     Cb = C + (long)bz * cStride;
  const float* Rb = residual ? (residual + (long)bz * rStride) : nullptr;

  int r  = lane & 15;              // row within A/B tile; C column
  int hl = lane >> 4;              // half select
  int am = tm * 16 + r;
  int bn = tn * 16 + r;
  int amc = (am < M) ? am : (M - 1);            // clamp: branch-free loads
  int bnc = (bn < N) ? bn : (N - 1);
  const h16* aptr = Ab + (long)amc * lda + (hl << 3);
  const h16* bptr = Bb + (long)bnc * ldb + (hl << 3);

  v8f acc = {};
  for (int k0 = 0; k0 < K; k0 += 32) {
    FragU af, bf;
    af.q[0] = *(const v4u*)(aptr);
    af.q[1] = *(const v4u*)(aptr + 16);
    bf.q[0] = *(const v4u*)(bptr);
    bf.q[1] = *(const v4u*)(bptr + 16);
    aptr += 32;
    bptr += 32;
    acc = __builtin_amdgcn_wmma_f32_16x16x32_f16(
        false, af.v, false, bf.v, (short)0, acc, false, false);
  }

#pragma unroll
  for (int i = 0; i < 8; ++i) {
    int mm = tm * 16 + hl * 8 + i;  // C row
    int nn = tn * 16 + r;           // C col
    if (mm < M && nn < N) {
      float v = acc[i];
      if (bias) v += bias[nn];
      if (Rb)   v += Rb[(long)mm * ldc + nn];
      if (doTanh) v = tanhf(v);
      long row = mm;
      if (remapT > 0) {             // m = b*T + t  ->  row = t*B + b
        int t = mm % remapT;
        int b = mm / remapT;
        row = (long)t * remapB + b;
      }
      Cb[row * (long)ldc + nn] = v;
    }
  }
}

// ---------------------------------------------------------------------------
// GRU recurrence: one workgroup per batch element; Whh^T resident in LDS
// (270KB of the 320KB WGP LDS -> recurrent weight read once from HBM).
// gi precomputed: gi[b][t][450].  h update per PyTorch GRU equations.
// ---------------------------------------------------------------------------
#define WLD 452
#define GRU_SMEM_FLOATS (150 * WLD + 152 + 452 + 450)

__global__ void gru_recurrence_kernel(
    const float* __restrict__ gi,   // [B][T][450]
    const float* __restrict__ Whh,  // [450][150]
    const float* __restrict__ bhh,  // [450]
    const float* __restrict__ h0,   // [B][150] or null
    float* __restrict__ ys,         // [B][T][150]
    float* __restrict__ hT,         // [B][150] or null
    int T)
{
  extern __shared__ float smem[];
  float* WhhT = smem;                    // [150][WLD]  (WhhT[k*WLD + j])
  float* hsh  = smem + 150 * WLD;        // [150] (+pad)
  float* gh   = hsh + 152;               // [450] (+pad)
  float* bsh  = gh + 452;                // [450]
  int b = blockIdx.x;
  int tid = threadIdx.x;
  int nt = blockDim.x;

  for (int idx = tid; idx < G3 * HH; idx += nt) {
    int j = idx / HH, k = idx - j * HH;
    WhhT[k * WLD + j] = Whh[idx];
  }
  for (int j = tid; j < G3; j += nt) bsh[j] = bhh[j];
  for (int i = tid; i < HH; i += nt)
    hsh[i] = h0 ? h0[(long)b * HH + i] : 0.0f;
  __syncthreads();

  const float* gib = gi + (long)b * T * G3;
  float* yb = ys + (long)b * T * HH;
  for (int t = 0; t < T; ++t) {
    const float* g = gib + (long)t * G3;
    for (int j = tid; j < G3; j += nt) {
      float acc = bsh[j];
      const float* wc = WhhT + j;
      for (int k = 0; k < HH; ++k) acc += hsh[k] * wc[k * WLD];
      gh[j] = acc;
    }
    __syncthreads();
    for (int i = tid; i < HH; i += nt) {
      float rr = 1.0f / (1.0f + expf(-(g[i] + gh[i])));
      float zz = 1.0f / (1.0f + expf(-(g[HH + i] + gh[HH + i])));
      float nn = tanhf(g[2 * HH + i] + rr * gh[2 * HH + i]);
      float hv = (1.0f - zz) * nn + zz * hsh[i];
      yb[(long)t * HH + i] = hv;
      hsh[i] = hv;
    }
    __syncthreads();
  }
  if (hT)
    for (int i = tid; i < HH; i += nt) hT[(long)b * HH + i] = hsh[i];
}

// ---------------------------------------------------------------------------
// Embedding gather -> f16, K-padded.  mode 0: token = idx[m]
// mode 1 (decoder input): m=b*T+t, token = (t==0) ? SOS : tgts[b*T+t-1]
// mode 2 (law):           m=b*T+l, token = law[charge_id[b]*T + l]
// ---------------------------------------------------------------------------
__global__ void gather_embed_kernel(
    const float* __restrict__ table, const int* __restrict__ idx,
    const int* __restrict__ idx2, h16* __restrict__ out,
    int M, int Ksrc, int Kpad, int mode, int Tlen)
{
  long g = (long)blockIdx.x * blockDim.x + threadIdx.x;
  long total = (long)M * Kpad;
  if (g >= total) return;
  int m = (int)(g / Kpad);
  int k = (int)(g - (long)m * Kpad);
  float v = 0.0f;
  if (k < Ksrc) {
    int token;
    if (mode == 0) token = idx[m];
    else if (mode == 1) {
      int t = m % Tlen, b = m / Tlen;
      token = (t == 0) ? SOS_TOK : idx[b * Tlen + t - 1];
    } else {
      int l = m % Tlen, b = m / Tlen;
      token = idx[idx2[b] * Tlen + l];
    }
    v = table[(long)token * Ksrc + k];
  }
  out[g] = (h16)v;
}

// f32 [M][Ksrc] -> f16 [M][Kpad] zero-padded
__global__ void cvt_f16_pad_kernel(const float* __restrict__ in,
                                   h16* __restrict__ out,
                                   int M, int Ksrc, int Kpad)
{
  long g = (long)blockIdx.x * blockDim.x + threadIdx.x;
  long total = (long)M * Kpad;
  if (g >= total) return;
  int m = (int)(g / Kpad);
  int k = (int)(g - (long)m * Kpad);
  out[g] = (h16)((k < Ksrc) ? in[(long)m * Ksrc + k] : 0.0f);
}

// f32 [B][R][C] -> f16 [B][C][R]
__global__ void transpose_f16_kernel(const float* __restrict__ in,
                                     h16* __restrict__ out,
                                     int B, int R, int C)
{
  long g = (long)blockIdx.x * blockDim.x + threadIdx.x;
  long total = (long)B * R * C;
  if (g >= total) return;
  int b = (int)(g / ((long)R * C));
  long rem = g - (long)b * R * C;
  int c = (int)(rem / R);
  int r = (int)(rem - (long)c * R);
  out[g] = (h16)in[((long)b * R + r) * C + c];
}

// mean over time: in [B][T][H] -> out [B][H]
__global__ void mean_time_kernel(const float* __restrict__ in,
                                 float* __restrict__ out, int T)
{
  int g = blockIdx.x * blockDim.x + threadIdx.x;
  if (g >= BZ * HH) return;
  int b = g / HH, i = g % HH;
  float s = 0.0f;
  for (int t = 0; t < T; ++t) s += in[((long)b * T + t) * HH + i];
  out[g] = s / (float)T;
}

// charge_out[b][c] = dot(hid[b], Wc[c]) + bc[c]
__global__ void charge_out_kernel(const float* __restrict__ hid,
                                  const float* __restrict__ Wc,
                                  const float* __restrict__ bc,
                                  float* __restrict__ out)
{
  int g = blockIdx.x * blockDim.x + threadIdx.x;
  if (g >= BZ * NC) return;
  int b = g / NC, c = g % NC;
  float s = bc[c];
  for (int k = 0; k < HH; ++k) s += hid[b * HH + k] * Wc[c * HH + k];
  out[g] = s;
}

__global__ void argmax_kernel(const float* __restrict__ co, int* __restrict__ id)
{
  int b = blockIdx.x * blockDim.x + threadIdx.x;
  if (b >= BZ) return;
  float best = co[b * NC]; int bi = 0;
  for (int c = 1; c < NC; ++c) {
    float v = co[b * NC + c];
    if (v > best) { best = v; bi = c; }
  }
  id[b] = bi;
}

// softmax with att==0 -> -inf mask and nan->0;  one thread per row, f16 out
__global__ void masked_softmax_kernel(const float* __restrict__ att,
                                      h16* __restrict__ aw, long rows, int L)
{
  long r = (long)blockIdx.x * blockDim.x + threadIdx.x;
  if (r >= rows) return;
  const float* a = att + r * L;
  h16* o = aw + r * L;
  float mx = -3.4e38f; bool any = false;
  for (int l = 0; l < L; ++l)
    if (a[l] != 0.0f) { any = true; mx = fmaxf(mx, a[l]); }
  float s = 0.0f;
  if (any) for (int l = 0; l < L; ++l)
    if (a[l] != 0.0f) s += expf(a[l] - mx);
  for (int l = 0; l < L; ++l) {
    float v = (any && a[l] != 0.0f) ? expf(a[l] - mx) / s : 0.0f;
    o[l] = (h16)v;
  }
}

// plain softmax, one thread per row, f16 out
__global__ void softmax_f16_kernel(const float* __restrict__ x,
                                   h16* __restrict__ y, long rows, int L)
{
  long r = (long)blockIdx.x * blockDim.x + threadIdx.x;
  if (r >= rows) return;
  const float* a = x + r * L;
  h16* o = y + r * L;
  float mx = -3.4e38f;
  for (int l = 0; l < L; ++l) mx = fmaxf(mx, a[l]);
  float s = 0.0f;
  for (int l = 0; l < L; ++l) s += expf(a[l] - mx);
  float inv = 1.0f / s;
  for (int l = 0; l < L; ++l) o[l] = (h16)(expf(a[l] - mx) * inv);
}

// concat [dec_out | context] f32 -> f16 [M][320] (pad 300..319 = 0)
__global__ void concat_kernel(const float* __restrict__ a,
                              const float* __restrict__ b2,
                              h16* __restrict__ out, int M)
{
  long g = (long)blockIdx.x * blockDim.x + threadIdx.x;
  long total = (long)M * 320;
  if (g >= total) return;
  int m = (int)(g / 320);
  int k = (int)(g - (long)m * 320);
  float v = 0.0f;
  if (k < HH) v = a[(long)m * HH + k];
  else if (k < 2 * HH) v = b2[(long)m * HH + (k - HH)];
  out[g] = (h16)v;
}

// per-(b,t) row NLL from logits stored [T][B][V] (i.e. d_out outs region)
__global__ void nll_kernel(const float* __restrict__ outs,
                           const int* __restrict__ tgts,
                           float* __restrict__ nll, int T, int Bn, int V)
{
  __shared__ float red[256];
  int rid = blockIdx.x;                 // rid = b*T + t
  int b = rid / T, t = rid % T;
  const float* row = outs + ((long)t * Bn + b) * (long)V;
  int tid = threadIdx.x;
  float mx = -3.4e38f;
  for (int v = tid; v < V; v += 256) mx = fmaxf(mx, row[v]);
  red[tid] = mx; __syncthreads();
  for (int s = 128; s > 0; s >>= 1) {
    if (tid < s) red[tid] = fmaxf(red[tid], red[tid + s]);
    __syncthreads();
  }
  mx = red[0]; __syncthreads();
  float sm = 0.0f;
  for (int v = tid; v < V; v += 256) sm += expf(row[v] - mx);
  red[tid] = sm; __syncthreads();
  for (int s = 128; s > 0; s >>= 1) {
    if (tid < s) red[tid] += red[tid + s];
    __syncthreads();
  }
  if (tid == 0) {
    int tg = tgts[b * T + t];
    nll[rid] = logf(red[0]) + mx - row[tg];
  }
}

// loss = sum_t ( sum_b nll*valid / max(sum_b valid,1) );  valid = tgt != EOS
__global__ void loss_kernel(const float* __restrict__ nll,
                            const int* __restrict__ tgts,
                            float* __restrict__ lossOut, int T, int Bn)
{
  __shared__ float red[256];
  int tid = threadIdx.x;
  float ps = 0.0f;
  for (int t = tid; t < T; t += 256) {
    float sn = 0.0f, cv = 0.0f;
    for (int b = 0; b < Bn; ++b) {
      int tok = tgts[b * T + t];
      float valid = (tok != EOS_TOK) ? 1.0f : 0.0f;
      sn += nll[b * T + t] * valid;
      cv += valid;
    }
    ps += sn / fmaxf(cv, 1.0f);
  }
  red[tid] = ps; __syncthreads();
  for (int s = 128; s > 0; s >>= 1) {
    if (tid < s) red[tid] += red[tid + s];
    __syncthreads();
  }
  if (tid == 0) *lossOut = red[0];
}

// ---------------------------------------------------------------------------
// Host orchestration
// ---------------------------------------------------------------------------
static inline unsigned cdiv(long a, long b) { return (unsigned)((a + b - 1) / b); }

extern "C" void kernel_launch(void* const* d_in, const int* in_sizes, int n_in,
                              void* d_out, int out_size, void* d_ws, size_t ws_size,
                              hipStream_t stream) {
  (void)in_sizes; (void)n_in; (void)out_size; (void)ws_size;
  const int*   srcs      = (const int*)d_in[0];
  const int*   tgts      = (const int*)d_in[1];
  const int*   law       = (const int*)d_in[2];
  const float* src_embed = (const float*)d_in[3];
  const float* e1Wih = (const float*)d_in[4];
  const float* e1Whh = (const float*)d_in[5];
  const float* e1bih = (const float*)d_in[6];
  const float* e1bhh = (const float*)d_in[7];
  const float* e2Wih = (const float*)d_in[8];
  const float* e2Whh = (const float*)d_in[9];
  const float* e2bih = (const float*)d_in[10];
  const float* e2bhh = (const float*)d_in[11];
  const float* dec_embed = (const float*)d_in[12];
  const float* dWih = (const float*)d_in[13];
  const float* dWhh = (const float*)d_in[14];
  const float* dbih = (const float*)d_in[15];
  const float* dbhh = (const float*)d_in[16];
  const float* Ww = (const float*)d_in[17];
  const float* bw = (const float*)d_in[18];
  const float* Wp = (const float*)d_in[19];
  const float* bp = (const float*)d_in[20];
  const float* Wc = (const float*)d_in[21];
  const float* bc = (const float*)d_in[22];
  float* out = (float*)d_out;

  const size_t OUTS = (size_t)TT * BZ * VT;   // 96,000,000 floats
  float* outLoss   = out + OUTS;
  float* outCharge = out + OUTS + 1;

  // ---- workspace layout (bump allocator, 256B aligned) --------------------
  char* ws = (char*)d_ws;
  size_t off = 0;
  auto alloc = [&](size_t bytes) -> char* {
    char* p = ws + off;
    off += (bytes + 255) & ~(size_t)255;
    return p;
  };
  h16*   W1_16   = (h16*)alloc((size_t)G3 * EP * 2);         // enc_rnn_Wih f16
  h16*   W2_16   = (h16*)alloc((size_t)G3 * HP * 2);         // enc_gru_Wih f16
  h16*   W3_16   = (h16*)alloc((size_t)G3 * EP * 2);         // dec_rnn_Wih f16
  h16*   WW16    = (h16*)alloc((size_t)HH * 320 * 2);        // Ww f16
  h16*   WP16    = (h16*)alloc((size_t)VT * HP * 2);         // Wp f16
  h16*   EMB16   = (h16*)alloc((size_t)BZ * LS * EP * 2);    // embed staging (reused)
  float* GI      = (float*)alloc((size_t)BZ * LS * G3 * 4);  // gate inputs (reused)
  float* ENC1    = (float*)alloc((size_t)BZ * LS * HH * 4);
  h16*   ENC1_16 = (h16*)alloc((size_t)BZ * LS * HP * 2);
  float* CH_HID  = (float*)alloc((size_t)BZ * HH * 4);
  int*   CH_ID   = (int*)alloc((size_t)BZ * 4);
  float* LEGALS  = (float*)alloc((size_t)BZ * LL * HH * 4);
  h16*   LEG16   = (h16*)alloc((size_t)BZ * LL * HP * 2);
  h16*   LEGT16  = (h16*)alloc((size_t)BZ * HH * LL * 2);
  float* ATT     = (float*)alloc((size_t)BZ * LS * LL * 4);
  h16*   AW16    = (h16*)alloc((size_t)BZ * LS * LL * 2);
  float* CAW     = (float*)alloc((size_t)BZ * LS * HH * 4);
  h16*   CAW16   = (h16*)alloc((size_t)BZ * LS * HP * 2);
  float* ENC2    = (float*)alloc((size_t)BZ * LS * HH * 4);
  float* HIDF    = (float*)alloc((size_t)BZ * HH * 4);
  h16*   ENC2_16 = (h16*)alloc((size_t)BZ * LS * HP * 2);
  h16*   ENC2T16 = (h16*)alloc((size_t)BZ * HH * LS * 2);
  float* DECOUT  = (float*)alloc((size_t)BZ * TT * HH * 4);
  h16*   DEC16   = (h16*)alloc((size_t)BZ * TT * HP * 2);
  float* ATT2    = (float*)alloc((size_t)BZ * TT * LS * 4);
  h16*   ATTW16  = (h16*)alloc((size_t)BZ * TT * LS * 2);
  float* CTX     = (float*)alloc((size_t)BZ * TT * HH * 4);
  h16*   CAT16   = (h16*)alloc((size_t)BZ * TT * 320 * 2);
  float* CO      = (float*)alloc((size_t)BZ * TT * HH * 4);
  h16*   CO16    = (h16*)alloc((size_t)BZ * TT * HP * 2);
  float* NLL     = (float*)alloc((size_t)BZ * TT * 4);

  const size_t gruSmem = (size_t)GRU_SMEM_FLOATS * 4;

  auto gemm = [&](const h16* A, const h16* Bm, const float* bias,
                  const float* resid, float* C, int M, int N, int K,
                  int lda, int ldb, int ldc,
                  long aS, long bS, long cS, long rS, int nb,
                  int doTanh, int remapT, int remapB) {
    dim3 g(cdiv(cdiv(N, 16), 4), cdiv(M, 16), nb);
    gemm_wmma_kernel<<<g, dim3(32, 4), 0, stream>>>(
        A, Bm, bias, resid, C, M, N, K, lda, ldb, ldc,
        aS, bS, cS, rS, doTanh, remapT, remapB);
  };
  auto cvt = [&](const float* in, h16* o, int M, int Ks, int Kp) {
    long tot = (long)M * Kp;
    cvt_f16_pad_kernel<<<cdiv(tot, 256), 256, 0, stream>>>(in, o, M, Ks, Kp);
  };

  // ---- phase 0: weight conversion to f16 (K-padded) -----------------------
  cvt(e1Wih, W1_16, G3, EM, EP);
  cvt(e2Wih, W2_16, G3, HH, HP);
  cvt(dWih,  W3_16, G3, EM, EP);
  cvt(Ww,    WW16,  HH, 2 * HH, 320);
  cvt(Wp,    WP16,  VT, HH, HP);

  // ---- phase 1: encoder pass 1 -------------------------------------------
  {
    long tot = (long)BZ * LS * EP;
    gather_embed_kernel<<<cdiv(tot, 256), 256, 0, stream>>>(
        src_embed, srcs, nullptr, EMB16, BZ * LS, EM, EP, 0, 0);
  }
  gemm(EMB16, W1_16, e1bih, nullptr, GI, BZ * LS, G3, EP, EP, EP, G3,
       0, 0, 0, 0, 1, 0, 0, 0);
  gru_recurrence_kernel<<<BZ, 256, gruSmem, stream>>>(
      GI, e1Whh, e1bhh, nullptr, ENC1, nullptr, LS);

  mean_time_kernel<<<cdiv(BZ * HH, 256), 256, 0, stream>>>(ENC1, CH_HID, LS);
  charge_out_kernel<<<cdiv(BZ * NC, 256), 256, 0, stream>>>(CH_HID, Wc, bc, outCharge);
  argmax_kernel<<<1, 64, 0, stream>>>(outCharge, CH_ID);

  // ---- phase 2: law-article encoding + mask attention + encoder pass 2 ----
  {
    long tot = (long)BZ * LL * EP;
    gather_embed_kernel<<<cdiv(tot, 256), 256, 0, stream>>>(
        src_embed, law, CH_ID, EMB16, BZ * LL, EM, EP, 2, LL);
  }
  gemm(EMB16, W1_16, e1bih, nullptr, GI, BZ * LL, G3, EP, EP, EP, G3,
       0, 0, 0, 0, 1, 0, 0, 0);
  gru_recurrence_kernel<<<BZ, 256, gruSmem, stream>>>(
      GI, e1Whh, e1bhh, nullptr, LEGALS, nullptr, LL);

  cvt(ENC1, ENC1_16, BZ * LS, HH, HP);
  cvt(LEGALS, LEG16, BZ * LL, HH, HP);
  {
    long tot = (long)BZ * HH * LL;
    transpose_f16_kernel<<<cdiv(tot, 256), 256, 0, stream>>>(LEGALS, LEGT16, BZ, LL, HH);
  }
  // att[b] = enc_outs1[b] (512xH) @ legals[b]^T (HxLl)
  gemm(ENC1_16, LEG16, nullptr, nullptr, ATT, LS, LL, HP, HP, HP, LL,
       (long)LS * HP, (long)LL * HP, (long)LS * LL, 0, BZ, 0, 0, 0);
  masked_softmax_kernel<<<cdiv((long)BZ * LS, 256), 256, 0, stream>>>(
      ATT, AW16, (long)BZ * LS, LL);
  // charge_aware[b] = aw[b] (512xLl) @ legals[b] (LlxH) + enc_outs1[b]
  gemm(AW16, LEGT16, nullptr, ENC1, CAW, LS, HH, LL, LL, LL, HH,
       (long)LS * LL, (long)HH * LL, (long)LS * HH, (long)LS * HH, BZ, 0, 0, 0);

  cvt(CAW, CAW16, BZ * LS, HH, HP);
  gemm(CAW16, W2_16, e2bih, nullptr, GI, BZ * LS, G3, HP, HP, HP, G3,
       0, 0, 0, 0, 1, 0, 0, 0);
  gru_recurrence_kernel<<<BZ, 256, gruSmem, stream>>>(
      GI, e2Whh, e2bhh, nullptr, ENC2, HIDF, LS);

  // ---- phase 3: decoder + attention + projection --------------------------
  {
    long tot = (long)BZ * TT * EP;
    gather_embed_kernel<<<cdiv(tot, 256), 256, 0, stream>>>(
        dec_embed, tgts, nullptr, EMB16, BZ * TT, EM, EP, 1, TT);
  }
  gemm(EMB16, W3_16, dbih, nullptr, GI, BZ * TT, G3, EP, EP, EP, G3,
       0, 0, 0, 0, 1, 0, 0, 0);
  gru_recurrence_kernel<<<BZ, 256, gruSmem, stream>>>(
      GI, dWhh, dbhh, HIDF, DECOUT, nullptr, TT);

  cvt(DECOUT, DEC16, BZ * TT, HH, HP);
  cvt(ENC2, ENC2_16, BZ * LS, HH, HP);
  {
    long tot = (long)BZ * HH * LS;
    transpose_f16_kernel<<<cdiv(tot, 256), 256, 0, stream>>>(ENC2, ENC2T16, BZ, LS, HH);
  }
  // att2[b] = dec_out[b] (TxH) @ enc_outs[b]^T (HxLs)
  gemm(DEC16, ENC2_16, nullptr, nullptr, ATT2, TT, LS, HP, HP, HP, LS,
       (long)TT * HP, (long)LS * HP, (long)TT * LS, 0, BZ, 0, 0, 0);
  softmax_f16_kernel<<<cdiv((long)BZ * TT, 256), 256, 0, stream>>>(
      ATT2, ATTW16, (long)BZ * TT, LS);
  // context[b] = attw[b] (TxLs) @ enc_outs[b] (LsxH)
  gemm(ATTW16, ENC2T16, nullptr, nullptr, CTX, TT, HH, LS, LS, LS, HH,
       (long)TT * LS, (long)HH * LS, (long)TT * HH, 0, BZ, 0, 0, 0);

  {
    long tot = (long)BZ * TT * 320;
    concat_kernel<<<cdiv(tot, 256), 256, 0, stream>>>(DECOUT, CTX, CAT16, BZ * TT);
  }
  // co = tanh(concat @ Ww^T + bw)
  gemm(CAT16, WW16, bw, nullptr, CO, BZ * TT, HH, 320, 320, 320, HH,
       0, 0, 0, 0, 1, 1, 0, 0);
  cvt(CO, CO16, BZ * TT, HH, HP);
  // logits -> d_out directly, transposed to [T][B][V] via row remap
  gemm(CO16, WP16, bp, nullptr, out, BZ * TT, VT, HP, HP, HP, VT,
       0, 0, 0, 0, 1, 0, TT, BZ);

  // ---- loss ---------------------------------------------------------------
  nll_kernel<<<BZ * TT, 256, 0, stream>>>(out, tgts, NLL, TT, BZ, VT);
  loss_kernel<<<1, 256, 0, stream>>>(NLL, tgts, outLoss, TT, BZ);
}
